// VAE_31499290149445
// MI455X (gfx1250) — matrix operand order repaired and checked by reading, not compiled, not run on hardware
//
#include <hip/hip_runtime.h>
#include <hip/hip_bf16.h>
#include <math.h>

// ---------------------------------------------------------------------------
// VAE ELBO forward, fused bf16 WMMA pipeline for gfx1250 (MI455X)
//   B=65536, D=784 (pad 832), H=1024, L=64, decoder N pad 896
// GEMMs use v_wmma_f32_16x16x32_bf16 with double-buffered LDS tiles filled by
// global_load_async_to_lds_b128 (ASYNCcnt-tracked async DMA, no staging VGPRs)
// ---------------------------------------------------------------------------

typedef __attribute__((ext_vector_type(16))) __bf16 v16bf;
typedef __attribute__((ext_vector_type(8)))  float  v8f;

union FragU { unsigned u[8]; v16bf v; };
static_assert(sizeof(v16bf) == 32, "v16bf must be 32 bytes");

__device__ __forceinline__ __bf16 f2bf(float f) {
  unsigned u = __builtin_bit_cast(unsigned, f);
  unsigned r = (u + 0x7FFFu + ((u >> 16) & 1u)) >> 16;   // round-to-nearest-even
  unsigned short s = (unsigned short)r;
  return __builtin_bit_cast(__bf16, s);
}

__device__ __forceinline__ float softplusf(float a) {
  return (a > 20.f) ? a : log1pf(__expf(a));
}

// Generic->LDS byte address: LDS aperture occupies the full upper 32 bits of
// the flat address, so addr[31:0] is the wave-relative LDS byte address.
__device__ __forceinline__ unsigned lds_u32(const void* p) {
  return (unsigned)(size_t)p;
}

// Async 16B copy global -> LDS (GV mode: 64-bit VGPR address, tracked by
// ASYNCcnt). vdst = LDS byte address VGPR.
__device__ __forceinline__ void async_copy_b128(unsigned lds_addr,
                                                const void* gptr) {
  asm volatile("global_load_async_to_lds_b128 %0, %1, off"
               :: "v"(lds_addr), "v"(gptr)
               : "memory");
}

// ---------------------------------------------------------------------------
// prep: convert inputs to bf16 (with K / N zero padding), build fused bias,
// zero the global scalar accumulator.
// ---------------------------------------------------------------------------
__global__ __launch_bounds__(256) void vae_prep_kernel(
    const float* __restrict__ x,   const float* __restrict__ We1,
    const float* __restrict__ Wmu, const float* __restrict__ Wls,
    const float* __restrict__ bmu, const float* __restrict__ bls,
    const float* __restrict__ Wd1, const float* __restrict__ Wd2,
    __bf16* __restrict__ xb,    __bf16* __restrict__ We1b,
    __bf16* __restrict__ Wmulsb, float* __restrict__ bmuls,
    __bf16* __restrict__ Wd1b,  __bf16* __restrict__ Wd2b,
    float* __restrict__ accum)
{
  const size_t gid    = (size_t)blockIdx.x * blockDim.x + threadIdx.x;
  const size_t stride = (size_t)gridDim.x * blockDim.x;
  if (gid == 0) *accum = 0.f;

  // x: [65536 x 784] -> [65536 x 832] bf16, zero-padded K
  for (size_t i = gid; i < (size_t)65536 * 832; i += stride) {
    size_t r = i / 832, c = i % 832;
    xb[i] = (c < 784) ? f2bf(x[r * 784 + c]) : f2bf(0.f);
  }
  // We1: [1024 x 784] -> [1024 x 832]
  for (size_t i = gid; i < (size_t)1024 * 832; i += stride) {
    size_t r = i / 832, c = i % 832;
    We1b[i] = (c < 784) ? f2bf(We1[r * 784 + c]) : f2bf(0.f);
  }
  // Wmu||Wls stacked: [128 x 1024]
  for (size_t i = gid; i < (size_t)128 * 1024; i += stride) {
    size_t r = i / 1024, c = i % 1024;
    Wmulsb[i] = f2bf((r < 64) ? Wmu[r * 1024 + c] : Wls[(r - 64) * 1024 + c]);
  }
  // bmu||bls: [128]
  for (size_t i = gid; i < 128; i += stride)
    bmuls[i] = (i < 64) ? bmu[i] : bls[i - 64];
  // Wd1: [1024 x 64]
  for (size_t i = gid; i < (size_t)1024 * 64; i += stride)
    Wd1b[i] = f2bf(Wd1[i]);
  // Wd2: [784 x 1024] -> [896 x 1024], zero-padded N rows
  for (size_t i = gid; i < (size_t)896 * 1024; i += stride) {
    size_t r = i / 1024, c = i % 1024;
    Wd2b[i] = (r < 784) ? f2bf(Wd2[r * 1024 + c]) : f2bf(0.f);
  }
}

// ---------------------------------------------------------------------------
// Block-tiled WMMA GEMM: D[m,n] = sum_k A[m,k] * W[n,k]  (both row-major bf16)
// Block tile 128x128, 8 waves, each wave owns a 32x64 sub-tile
// (2 x 4 f32 accumulators). K stepped by 64 through double-buffered LDS tiles
// filled with async b128 loads; 16 v_wmma per K-step.
//
// MODE 0: out = tanh(D + bias)          -> bf16 store (ldo)
// MODE 1: out = D + bias                -> f32  store (ldo)
// MODE 3: bce: a = D + bias; accumulate x*log(sig(a)) + (1-x)*log(1-sig(a))
//         (n masked to < 784), block-reduced, atomicAdd into *accum
// ---------------------------------------------------------------------------
template <int MODE>
__global__ __launch_bounds__(256) void vae_gemm_kernel(
    const __bf16* __restrict__ A, int lda,
    const __bf16* __restrict__ W, int ldw,
    int K,
    const float* __restrict__ bias,
    void* __restrict__ outp, int ldo,
    const float* __restrict__ xin,
    float* __restrict__ accum)
{
  __shared__ __align__(16) __bf16 As[2][128 * 64];
  __shared__ __align__(16) __bf16 Bs[2][128 * 64];

  const int tid  = threadIdx.x;
  const int lane = tid & 31;
  const int wid  = tid >> 5;
  const int wm   = wid & 3;        // 4 waves along M -> 4*32 = 128
  const int wn   = wid >> 2;       // 2 waves along N -> 2*64 = 128
  const int m0   = blockIdx.x * 128;
  const int n0   = blockIdx.y * 128;

  // Issue one 128x64 bf16 tile pair (A and W) as async b128 DMA into LDS
  // buffer p. 4+4 instructions per thread -> 8 outstanding per wave.
  auto issue_tile = [&](int kk, int p) {
    const int k0 = kk << 6;
#pragma unroll
    for (int i = 0; i < 4; ++i) {
      const int c   = tid + (i << 8);
      const int row = c >> 3;
      const int col = (c & 7) << 3;
      async_copy_b128(lds_u32(&As[p][row * 64 + col]),
                      A + (size_t)(m0 + row) * lda + k0 + col);
      async_copy_b128(lds_u32(&Bs[p][row * 64 + col]),
                      W + (size_t)(n0 + row) * ldw + k0 + col);
    }
  };

  const v8f vzero = {0.f, 0.f, 0.f, 0.f, 0.f, 0.f, 0.f, 0.f};
  v8f acc[2][4];
#pragma unroll
  for (int am = 0; am < 2; ++am)
#pragma unroll
    for (int bn = 0; bn < 4; ++bn) acc[am][bn] = vzero;

  const int ksteps = K >> 6;
  issue_tile(0, 0);
  for (int kk = 0; kk < ksteps; ++kk) {
    const int p = kk & 1;
    if (kk + 1 < ksteps) {
      issue_tile(kk + 1, p ^ 1);
      // leave only the 8 just-issued loads in flight: buffer p is complete
      asm volatile("s_wait_asynccnt 0x8" ::: "memory");
    } else {
      asm volatile("s_wait_asynccnt 0x0" ::: "memory");
    }
    __syncthreads();

#pragma unroll
    for (int ks = 0; ks < 64; ks += 32) {
      // A fragment (16x32 bf16): lanes 0-15 => M rows, K 0..7 & 16..23;
      // lanes 16-31 => same M rows, K 8..15 & 24..31. 2 bf16 per VGPR.
      v16bf afrag[2];
#pragma unroll
      for (int am = 0; am < 2; ++am) {
        const int m  = wm * 32 + am * 16 + (lane & 15);
        const int kh = (lane >> 4) << 3;
        FragU f;
#pragma unroll
        for (int v = 0; v < 8; ++v) {
          const int k = ((v >> 2) << 4) + kh + ((v & 3) << 1);
          f.u[v] = *(const unsigned*)(&As[p][m * 64 + ks + k]);
        }
        afrag[am] = f.v;
      }
      // B fragment (32x16, B[k][n] = W[n][k]): lanes 0-15 -> K 0..7 & 16..23,
      // lanes 16-31 -> K 8..15 & 24..31; N = lane%16.
      v16bf bfrag[4];
#pragma unroll
      for (int bn = 0; bn < 4; ++bn) {
        const int n  = wn * 64 + bn * 16 + (lane & 15);
        const int kb = (lane >> 4) << 3;
        FragU f;
#pragma unroll
        for (int v = 0; v < 8; ++v)
          f.u[v] = *(const unsigned*)(&Bs[p][n * 64 + ks + ((v >> 2) << 4) + kb + ((v & 3) << 1)]);
        bfrag[bn] = f.v;
      }

#pragma unroll
      for (int am = 0; am < 2; ++am)
#pragma unroll
        for (int bn = 0; bn < 4; ++bn)
          acc[am][bn] = __builtin_amdgcn_wmma_f32_16x16x32_bf16(
              false, afrag[am], false, bfrag[bn], (short)0, acc[am][bn],
              false, false);
    }
    __syncthreads();
  }

  // Epilogue. C/D layout: VGPR v holds M = v (lanes 0-15) / v+8 (lanes 16-31),
  // N = lane % 16.
  float lsum = 0.f;
#pragma unroll
  for (int am = 0; am < 2; ++am) {
#pragma unroll
    for (int bn = 0; bn < 4; ++bn) {
#pragma unroll
      for (int v = 0; v < 8; ++v) {
        const int m   = m0 + wm * 32 + am * 16 + v + ((lane >> 4) << 3);
        const int n   = n0 + wn * 64 + bn * 16 + (lane & 15);
        const float d = acc[am][bn][v];
        if constexpr (MODE == 0) {
          ((__bf16*)outp)[(size_t)m * ldo + n] = f2bf(tanhf(d + bias[n]));
        } else if constexpr (MODE == 1) {
          ((float*)outp)[(size_t)m * ldo + n] = d + bias[n];
        } else {
          if (n < 784) {
            const float a  = d + bias[n];
            const float xv = xin[(size_t)m * 784 + n];
            // x*log(sigmoid(a)) + (1-x)*log(1-sigmoid(a)), numerically stable
            lsum -= xv * softplusf(-a) + (1.f - xv) * softplusf(a);
          }
        }
      }
    }
  }
  if constexpr (MODE == 3) {
    __shared__ float red[256];
    red[tid] = lsum;
    __syncthreads();
#pragma unroll
    for (int s = 128; s > 0; s >>= 1) {
      if (tid < s) red[tid] += red[tid + s];
      __syncthreads();
    }
    if (tid == 0) atomicAdd(accum, red[0]);
  }
}

// ---------------------------------------------------------------------------
// Reparameterization + Gaussian log-pdf row terms.
// row term = -log_q + log_p_z; the ±0.5*L*LOG2PI parts cancel analytically:
//   = 0.5*sum(log sig) + 0.5*(sum sig*eps^2)*(sum sig) - 0.5*L*sum z^2
// Also emits z as bf16 for the decoder GEMM.
// ---------------------------------------------------------------------------
__global__ __launch_bounds__(256) void vae_reparam_kernel(
    const float* __restrict__ muls, const float* __restrict__ eps,
    __bf16* __restrict__ zb, float* __restrict__ accum)
{
  const int n = blockIdx.x * blockDim.x + threadIdx.x;   // batch row
  const float* row = muls + (size_t)n * 128;
  float slog = 0.f, ssig = 0.f, sd2 = 0.f, sz2 = 0.f;
#pragma unroll 8
  for (int j = 0; j < 64; ++j) {
    const float mu  = row[j];
    const float ls  = row[64 + j];
    const float sig = __expf(ls);
    const float e   = eps[(size_t)n * 64 + j];
    const float z   = mu + sqrtf(sig) * e;
    zb[(size_t)n * 64 + j] = f2bf(z);
    slog += ls;                 // log(exp(ls)) == ls
    ssig += sig;
    sd2  += sig * e * e;        // (z-mu)^2 = sig*eps^2
    sz2  += z * z;
  }
  float t = 0.5f * slog + 0.5f * sd2 * ssig - 32.f * sz2;

  __shared__ float red[256];
  red[threadIdx.x] = t;
  __syncthreads();
#pragma unroll
  for (int s = 128; s > 0; s >>= 1) {
    if (threadIdx.x < s) red[threadIdx.x] += red[threadIdx.x + s];
    __syncthreads();
  }
  if (threadIdx.x == 0) atomicAdd(accum, red[0]);
}

__global__ void vae_final_kernel(const float* __restrict__ accum,
                                 float* __restrict__ out) {
  if (threadIdx.x == 0 && blockIdx.x == 0)
    out[0] = accum[0] * (1.0f / 65536.0f);
}

// ---------------------------------------------------------------------------
extern "C" void kernel_launch(void* const* d_in, const int* in_sizes, int n_in,
                              void* d_out, int out_size, void* d_ws,
                              size_t ws_size, hipStream_t stream) {
  constexpr int B = 65536, Dp = 832, H = 1024, L = 64, N4 = 896;

  const float* x   = (const float*)d_in[0];
  const float* eps = (const float*)d_in[1];
  const float* We1 = (const float*)d_in[2];
  const float* be1 = (const float*)d_in[3];
  const float* Wmu = (const float*)d_in[4];
  const float* bmu = (const float*)d_in[5];
  const float* Wls = (const float*)d_in[6];
  const float* bls = (const float*)d_in[7];
  const float* Wd1 = (const float*)d_in[8];
  const float* bd1 = (const float*)d_in[9];
  const float* Wd2 = (const float*)d_in[10];
  const float* bd2 = (const float*)d_in[11];
  float* out = (float*)d_out;

  char* ws = (char*)d_ws;
  size_t off = 0;
  auto take = [&](size_t bytes) -> void* {
    void* p = ws + off;
    off += (bytes + 255) & ~(size_t)255;
    return p;
  };
  float*  accum  = (float*) take(sizeof(float));
  __bf16* xb     = (__bf16*)take((size_t)B * Dp * 2);     // 109 MB
  __bf16* We1b   = (__bf16*)take((size_t)H * Dp * 2);
  __bf16* Wmulsb = (__bf16*)take((size_t)128 * H * 2);
  float*  bmuls  = (float*) take((size_t)128 * 4);
  __bf16* Wd1b   = (__bf16*)take((size_t)H * L * 2);
  __bf16* Wd2b   = (__bf16*)take((size_t)N4 * H * 2);
  __bf16* hb     = (__bf16*)take((size_t)B * H * 2);      // 134 MB
  float*  muls   = (float*) take((size_t)B * 128 * 4);    // 34 MB
  __bf16* zb     = (__bf16*)take((size_t)B * L * 2);
  __bf16* hdb    = (__bf16*)take((size_t)B * H * 2);      // 134 MB
  (void)ws_size; (void)in_sizes; (void)n_in; (void)out_size;

  // 1) bf16 conversion + padding + accumulator zero
  vae_prep_kernel<<<4096, 256, 0, stream>>>(x, We1, Wmu, Wls, bmu, bls, Wd1,
                                            Wd2, xb, We1b, Wmulsb, bmuls, Wd1b,
                                            Wd2b, accum);
  // 2) h = tanh(x @ We1^T + be1): [B x 1024], K=832
  vae_gemm_kernel<0><<<dim3(B / 128, H / 128), 256, 0, stream>>>(
      xb, Dp, We1b, Dp, Dp, be1, hb, H, nullptr, nullptr);
  // 3) [mu || log sig^2] = h @ [Wmu;Wls]^T + [bmu;bls]: [B x 128], K=1024
  vae_gemm_kernel<1><<<dim3(B / 128, 1), 256, 0, stream>>>(
      hb, H, Wmulsb, H, H, bmuls, muls, 128, nullptr, nullptr);
  // 4) z = mu + sqrt(sig)*eps; row Gaussian log-pdf terms -> accum
  vae_reparam_kernel<<<B / 256, 256, 0, stream>>>(muls, eps, zb, accum);
  // 5) hd = tanh(z @ Wd1^T + bd1): [B x 1024], K=64
  vae_gemm_kernel<0><<<dim3(B / 128, H / 128), 256, 0, stream>>>(
      zb, L, Wd1b, L, L, bd1, hdb, H, nullptr, nullptr);
  // 6) Bernoulli log-likelihood of sigmoid(hd @ Wd2^T + bd2) vs x -> accum
  vae_gemm_kernel<3><<<dim3(B / 128, N4 / 128), 256, 0, stream>>>(
      hdb, H, Wd2b, H, H, bd2, nullptr, 0, x, accum);
  // 7) elbo = accum / B
  vae_final_kernel<<<1, 64, 0, stream>>>(accum, out);
}
